// SelfAttention_11493332484800
// MI455X (gfx1250) — matrix operand order repaired and checked by reading, not compiled, not run on hardware
//
#include <hip/hip_runtime.h>
#include <hip/hip_bf16.h>

// ---------------------------------------------------------------------------
// Self-attention forward for MI455X (gfx1250).
//   bf16 WMMA GEMMs, double-buffered async HBM->LDS staging
//   (global_load_async_to_lds_b128 / s_wait_asynccnt), wave tile 32x64.
// ---------------------------------------------------------------------------

typedef __bf16 bf16_t;
typedef __attribute__((ext_vector_type(16))) __bf16 v16bf;
typedef __attribute__((ext_vector_type(8)))  float  v8f;

#define ATTN_NEG_INF -1000000000.0f

__device__ __forceinline__ bf16_t f2bf(float f) { return (bf16_t)f; }

// flat shared-aperture address -> 32-bit LDS byte offset (ISA: addr[31:0])
__device__ __forceinline__ unsigned lds_off(const void* p) {
  return (unsigned)(unsigned long long)p;
}

// async 16B copy: global -> LDS (tracked with ASYNCcnt)
__device__ __forceinline__ void async_copy_b128(unsigned dst_lds, const void* src) {
  asm volatile("global_load_async_to_lds_b128 %0, %1, off"
               :: "v"(dst_lds), "v"(src) : "memory");
}

// ---------------- elementwise f32 -> bf16 ----------------------------------
__global__ void convert_f32_bf16(const float* __restrict__ in,
                                 bf16_t* __restrict__ out, int n) {
  int i = blockIdx.x * blockDim.x + threadIdx.x;
  if (i < n) out[i] = f2bf(in[i]);
}

// ---------------- transpose + convert DxD weight ---------------------------
__global__ void transpose_convert(const float* __restrict__ in,
                                  bf16_t* __restrict__ out, int D) {
  int e = blockIdx.x * blockDim.x + threadIdx.x;
  int d = blockIdx.y * blockDim.y + threadIdx.y;
  if (e < D && d < D) out[(size_t)e * D + d] = f2bf(in[(size_t)d * D + e]);
}

// ---------------- generic C = A * B^T WMMA GEMM ----------------------------
// A  : M x K row-major bf16 ; Bt : N x K row-major bf16.
// Block tile 256(M) x 64(N), K stepped by 32. 8 waves; wave w owns a 32-row
// strip (2 A sub-tiles x 4 B sub-tiles = 8 WMMA per K step).
// modes: 0 = f32 *scale | 1 = bf16 | 2 = bf16 transposed (v^T) | 3 = f32 *mask
union FragAB { v16bf v; uint4 u[2]; };

__global__ void __launch_bounds__(256)
gemm_bt_wmma(const bf16_t* __restrict__ A, const bf16_t* __restrict__ Bt,
             void* __restrict__ Cout, const float* __restrict__ mask,
             int M, int N, int K,
             long long strideA, long long strideB, long long strideC,
             float scale, int mode, int S) {
  __shared__ __align__(16) bf16_t sA[2][256 * 32];   // 16KB per buffer
  __shared__ __align__(16) bf16_t sB[2][64 * 32];    // 4KB per buffer

  const int tid  = threadIdx.x;
  const int lane = tid & 31;
  const int wave = tid >> 5;
  const int z    = blockIdx.z;

  const int mBlock = blockIdx.y * 256;
  const int nBlock = blockIdx.x * 64;

  const bf16_t* Ab = A  + (size_t)z * strideA;
  const bf16_t* Bb = Bt + (size_t)z * strideB;

  const int halfSel = (lane & 16) ? 1 : 0;
  const int aoff    = halfSel ? 8 : 0;     // A: K sub-offset per half-wave
  const int boff    = halfSel ? 8 : 0;     // B: K sub-offset per half-wave
  const int lane15  = lane & 15;

  // staging decomposition: 16B chunk c -> row = c/4, col8 = (c%4)*8 elements
  const int cRow = tid >> 2, cCol = (tid & 3) * 8;

  v8f acc[2][4];
#pragma unroll
  for (int i = 0; i < 2; ++i)
#pragma unroll
    for (int j = 0; j < 4; ++j)
      acc[i][j] = (v8f){0.f,0.f,0.f,0.f,0.f,0.f,0.f,0.f};

  // ---- stage k0 = 0 into buffer 0 (5 async ops per thread per stage:
  //      A tile 256x32 = 1024 chunks (4/thread), B tile 64x32 = 256 (1/thread))
#pragma unroll
  for (int p = 0; p < 4; ++p) {
    async_copy_b128(lds_off(&sA[0][(cRow + 64 * p) * 32 + cCol]),
                    Ab + (size_t)(mBlock + cRow + 64 * p) * K + cCol);
  }
  async_copy_b128(lds_off(&sB[0][cRow * 32 + cCol]),
                  Bb + (size_t)(nBlock + cRow) * K + cCol);

  for (int k0 = 0; k0 < K; k0 += 32) {
    const int  cur     = (k0 >> 5) & 1;
    const int  nxt     = cur ^ 1;
    const bool hasNext = (k0 + 32) < K;

    if (hasNext) {  // stage next K tile into the other buffer
      const int kn = k0 + 32;
#pragma unroll
      for (int p = 0; p < 4; ++p) {
        async_copy_b128(lds_off(&sA[nxt][(cRow + 64 * p) * 32 + cCol]),
                        Ab + (size_t)(mBlock + cRow + 64 * p) * K + kn + cCol);
      }
      async_copy_b128(lds_off(&sB[nxt][cRow * 32 + cCol]),
                      Bb + (size_t)(nBlock + cRow) * K + kn + cCol);
      asm volatile("s_wait_asynccnt 0x5" ::: "memory");  // current stage done
    } else {
      asm volatile("s_wait_asynccnt 0x0" ::: "memory");
    }
    __syncthreads();   // all waves' copies of `cur` complete

    // ---- fragments from LDS (ds_load_b128) ----
    FragAB fa[2];
#pragma unroll
    for (int i = 0; i < 2; ++i) {
      const bf16_t* sa = &sA[cur][(wave * 32 + i * 16 + lane15) * 32];
      fa[i].u[0] = *(const uint4*)(sa + aoff);
      fa[i].u[1] = *(const uint4*)(sa + 16 + aoff);
    }
    FragAB fb[4];
#pragma unroll
    for (int j = 0; j < 4; ++j) {
      const bf16_t* sb = &sB[cur][(j * 16 + lane15) * 32 + boff];
      fb[j].u[0] = *(const uint4*)(sb);
      fb[j].u[1] = *(const uint4*)(sb + 16);
    }

#pragma unroll
    for (int i = 0; i < 2; ++i)
#pragma unroll
      for (int j = 0; j < 4; ++j) {
        acc[i][j] = __builtin_amdgcn_wmma_f32_16x16x32_bf16(
            false, fa[i].v, false, fb[j].v, (short)0, acc[i][j], false, false);
      }
    __syncthreads();   // done reading `cur` before it is overwritten
  }

  // ---- epilogue ----
  const int r0 = halfSel ? 8 : 0;
  if (mode == 0) {
    float* C = (float*)Cout + (size_t)z * strideC;
#pragma unroll
    for (int i = 0; i < 2; ++i)
#pragma unroll
      for (int j = 0; j < 4; ++j)
#pragma unroll
        for (int r = 0; r < 8; ++r) {
          int row = mBlock + wave * 32 + i * 16 + r0 + r;
          int col = nBlock + j * 16 + lane15;
          C[(size_t)row * N + col] = acc[i][j][r] * scale;
        }
  } else if (mode == 1) {
    bf16_t* C = (bf16_t*)Cout + (size_t)z * strideC;
#pragma unroll
    for (int i = 0; i < 2; ++i)
#pragma unroll
      for (int j = 0; j < 4; ++j)
#pragma unroll
        for (int r = 0; r < 8; ++r) {
          int row = mBlock + wave * 32 + i * 16 + r0 + r;
          int col = nBlock + j * 16 + lane15;
          C[(size_t)row * N + col] = f2bf(acc[i][j][r]);
        }
  } else if (mode == 2) {
    bf16_t* C = (bf16_t*)Cout;
#pragma unroll
    for (int i = 0; i < 2; ++i)
#pragma unroll
      for (int j = 0; j < 4; ++j)
#pragma unroll
        for (int r = 0; r < 8; ++r) {
          int row = mBlock + wave * 32 + i * 16 + r0 + r;
          int col = nBlock + j * 16 + lane15;
          int b = row / S, sr = row - b * S;
          C[((size_t)b * N + col) * S + sr] = f2bf(acc[i][j][r]);
        }
  } else { // mode 3
    float* C = (float*)Cout + (size_t)z * strideC;
    const float* mrow = mask + (size_t)z * S;
#pragma unroll
    for (int i = 0; i < 2; ++i)
#pragma unroll
      for (int j = 0; j < 4; ++j)
#pragma unroll
        for (int r = 0; r < 8; ++r) {
          int row = mBlock + wave * 32 + i * 16 + r0 + r;
          int col = nBlock + j * 16 + lane15;
          C[(size_t)row * N + col] = acc[i][j][r] * mrow[row];
        }
  }
}

// ---------------- row softmax (in place) + bf16 copy -----------------------
__global__ void __launch_bounds__(256)
softmax_rows(float* __restrict__ logits, bf16_t* __restrict__ P,
             const float* __restrict__ mask, int S) {
  const int row = blockIdx.x;          // 0 .. B*S-1
  const int b   = row / S;
  float*        l    = logits + (size_t)row * S;
  bf16_t*       p    = P      + (size_t)row * S;
  const float*  mrow = mask   + (size_t)b * S;
  __shared__ float sred[256];
  const int t = threadIdx.x;

  float mx = -3.4e38f;
  for (int k = t; k < S; k += 256) {
    float v = l[k] + (1.0f - mrow[k]) * ATTN_NEG_INF;
    mx = fmaxf(mx, v);
  }
  sred[t] = mx; __syncthreads();
  for (int s = 128; s > 0; s >>= 1) {
    if (t < s) sred[t] = fmaxf(sred[t], sred[t + s]);
    __syncthreads();
  }
  mx = sred[0]; __syncthreads();

  float sum = 0.f;
  for (int k = t; k < S; k += 256) {
    float v = __expf(l[k] + (1.0f - mrow[k]) * ATTN_NEG_INF - mx);
    l[k] = v;
    sum += v;
  }
  sred[t] = sum; __syncthreads();
  for (int s = 128; s > 0; s >>= 1) {
    if (t < s) sred[t] += sred[t + s];
    __syncthreads();
  }
  const float inv = 1.0f / sred[0];

  for (int k = t; k < S; k += 256) {
    float v = l[k] * inv;
    l[k] = v;
    p[k] = f2bf(v);
  }
}

// ---------------------------------------------------------------------------
extern "C" void kernel_launch(void* const* d_in, const int* in_sizes, int n_in,
                              void* d_out, int out_size, void* d_ws, size_t ws_size,
                              hipStream_t stream) {
  const float* x    = (const float*)d_in[0];   // (8, 2048, 512)
  const float* mask = (const float*)d_in[1];   // (8, 2048)
  const float* Qw   = (const float*)d_in[2];   // (512, 512)
  const float* Kw   = (const float*)d_in[3];
  const float* Vw   = (const float*)d_in[4];

  const int B = 8, S = 2048, D = 512;
  const size_t MS = (size_t)B * S;             // 16384

  float* out  = (float*)d_out;                 // (B,S,D)
  float* attn = out + MS * D;                  // (B,S,S)

  char* ws = (char*)d_ws;
  bf16_t* xb = (bf16_t*)ws;  ws += MS * D * sizeof(bf16_t);
  bf16_t* Qt = (bf16_t*)ws;  ws += (size_t)D * D * sizeof(bf16_t);
  bf16_t* Kt = (bf16_t*)ws;  ws += (size_t)D * D * sizeof(bf16_t);
  bf16_t* Vt = (bf16_t*)ws;  ws += (size_t)D * D * sizeof(bf16_t);
  bf16_t* qb = (bf16_t*)ws;  ws += MS * D * sizeof(bf16_t);
  bf16_t* kb = (bf16_t*)ws;  ws += MS * D * sizeof(bf16_t);
  bf16_t* vT = (bf16_t*)ws;  ws += MS * D * sizeof(bf16_t);   // (B, D, S)
  bf16_t* P  = (bf16_t*)ws;                                    // (B, S, S)

  {
    int n = (int)(MS * D);
    convert_f32_bf16<<<(n + 255) / 256, 256, 0, stream>>>(x, xb, n);
  }
  {
    dim3 blk(16, 16), grd(D / 16, D / 16);
    transpose_convert<<<grd, blk, 0, stream>>>(Qw, Qt, D);
    transpose_convert<<<grd, blk, 0, stream>>>(Kw, Kt, D);
    transpose_convert<<<grd, blk, 0, stream>>>(Vw, Vt, D);
  }
  {
    dim3 blk(256), grd(D / 64, (unsigned)(MS / 256), 1);
    gemm_bt_wmma<<<grd, blk, 0, stream>>>(xb, Qt, qb, nullptr,
        (int)MS, D, D, 0, 0, 0, 1.0f, /*mode=*/1, S);
    gemm_bt_wmma<<<grd, blk, 0, stream>>>(xb, Kt, kb, nullptr,
        (int)MS, D, D, 0, 0, 0, 1.0f, /*mode=*/1, S);
    gemm_bt_wmma<<<grd, blk, 0, stream>>>(xb, Vt, vT, nullptr,
        (int)MS, D, D, 0, 0, 0, 1.0f, /*mode=*/2, S);
  }
  {
    const float scale = 1.0f / sqrtf((float)D);
    dim3 blk(256), grd(S / 64, S / 256, B);
    gemm_bt_wmma<<<grd, blk, 0, stream>>>(qb, kb, attn, nullptr,
        S, S, D, (long long)S * D, (long long)S * D, (long long)S * S,
        scale, /*mode=*/0, S);
  }
  softmax_rows<<<(unsigned)MS, 256, 0, stream>>>(attn, P, mask, S);
  {
    dim3 blk(256), grd(D / 64, S / 256, B);
    gemm_bt_wmma<<<grd, blk, 0, stream>>>(P, vT, out, mask,
        S, D, S, (long long)S * S, (long long)D * S, (long long)S * D,
        1.0f, /*mode=*/3, S);
  }
}
